// GraphDecoder_85667417686140
// MI455X (gfx1250) — matrix operand order repaired and checked by reading, not compiled, not run on hardware
//
#include <hip/hip_runtime.h>
#include <hip/hip_bf16.h>

#define NUM_IN    64
#define BATCH_    32768
#define N_TOTAL_  393216            // BATCH_ * 12
#define N_EDGES_  1572864           // 4 * N_TOTAL_

typedef __bf16 bf16_t;
typedef bf16_t v16bf __attribute__((ext_vector_type(16)));
typedef float  v8f   __attribute__((ext_vector_type(8)));

static __device__ __forceinline__ unsigned short f2bf(float f) {
  unsigned int u = __float_as_uint(f);
  u += 0x7FFFu + ((u >> 16) & 1u);          // round-to-nearest-even
  return (unsigned short)(u >> 16);
}
static __device__ __forceinline__ float bf2f(unsigned short s) {
  return __uint_as_float(((unsigned int)s) << 16);
}

union FragAB { v16bf v; unsigned int u[8]; };
union FragC  { v8f   v; float        f[8]; };

// ---------------------------------------------------------------------------
// Generic WMMA GEMM: C[M,N] = act(A[M,K](bf16) * Bt[N,K](bf16)^T + bias)
// One wave computes one 16x16 tile; K stepped by 32 with
// v_wmma_f32_16x16x32_bf16. Fragment packing uses contiguous k order, same for A and B:
//   A 16x32: lane<16 row M=l16 holds K = kb+0..15 (pairs per VGPR),
//            lane>=16 holds K = kb+16..31
//   B 32x16: lane<16 col N=l16 holds K = kb+0..15, lane>=16 holds kb+16..31
//   C/D:     element r -> row mBase + 8*half + r, col nBase + l16
// ---------------------------------------------------------------------------
__global__ __launch_bounds__(128) void wmma_gemm(
    const unsigned short* __restrict__ A,
    const unsigned short* __restrict__ Bt,
    const float* __restrict__ bias,
    float* __restrict__ Cf,
    unsigned short* __restrict__ Cb,
    int M, int N, int K, int relu)
{
  const int wave   = blockIdx.x * (blockDim.x >> 5) + (threadIdx.x >> 5);
  const int tilesN = N >> 4;
  const int tm     = wave / tilesN;
  const int tn     = wave - tm * tilesN;
  const int mBase  = tm << 4, nBase = tn << 4;
  const int lane   = threadIdx.x & 31;
  const int hlf    = lane >> 4;
  const int l16    = lane & 15;

  FragC c;
#pragma unroll
  for (int r = 0; r < 8; ++r) c.f[r] = 0.0f;

  const unsigned short* Arow = A  + (long long)(mBase + l16) * K;
  const unsigned short* Bcol = Bt + (long long)(nBase + l16) * K;

  for (int kb = 0; kb < K; kb += 32) {
    FragAB a, b;
#pragma unroll
    for (int p = 0; p < 8; ++p) {
      const int ka = kb + 16 * hlf + 2 * p;
      a.u[p] = *(const unsigned int*)(Arow + ka);
      const int kbK = kb + 16 * hlf + 2 * p;
      b.u[p] = *(const unsigned int*)(Bcol + kbK);
    }
    c.v = __builtin_amdgcn_wmma_f32_16x16x32_bf16(
        /*neg_a=*/false, a.v, /*neg_b=*/false, b.v,
        /*c_mod=*/(short)0, c.v, /*reuse_a=*/false, /*reuse_b=*/false);
  }

  const int col = nBase + l16;
  const float bv = bias ? bias[col] : 0.0f;
#pragma unroll
  for (int r = 0; r < 8; ++r) {
    const long long row = mBase + 8 * hlf + r;
    float v = c.f[r] + bv;
    if (relu) v = fmaxf(v, 0.0f);
    if (Cb) Cb[row * (long long)N + col] = f2bf(v);
    else    Cf[row * (long long)N + col] = v;
  }
}

// --------------------------- elementwise helpers ---------------------------
__global__ void k_f32_to_bf16(const float* __restrict__ in,
                              unsigned short* __restrict__ out, long long n) {
  long long i = (long long)blockIdx.x * blockDim.x + threadIdx.x;
  if (i < n) out[i] = f2bf(in[i]);
}

// W[K,N] f32 -> Wt[N,K] bf16
__global__ void k_transpose_bf16(const float* __restrict__ W,
                                 unsigned short* __restrict__ Wt, int K, int N) {
  int i = blockIdx.x * blockDim.x + threadIdx.x;
  if (i < K * N) {
    int k = i / N, n = i - k * N;
    Wt[(long long)n * K + k] = f2bf(W[i]);
  }
}

__global__ void k_init_deg(float* __restrict__ deg, int n) {
  int i = blockIdx.x * blockDim.x + threadIdx.x;
  if (i < n) deg[i] = 1.0f;                      // self-loop contribution
}

__global__ void k_deg(const int* __restrict__ cols, float* __restrict__ deg,
                      long long E) {
  long long e = (long long)blockIdx.x * blockDim.x + threadIdx.x;
  if (e < E) atomicAdd(&deg[cols[e]], 1.0f);
}

__global__ void k_dinv(float* __restrict__ deg, int n) {
  int i = blockIdx.x * blockDim.x + threadIdx.x;
  if (i < n) deg[i] = rsqrtf(deg[i]);            // deg >= 1 always
}

// scatter-add of bf16 messages (conv1): 4 features per thread
__global__ void k_scatter_bf16(const int* __restrict__ rows,
                               const int* __restrict__ cols,
                               const unsigned short* __restrict__ X,
                               const float* __restrict__ dinv,
                               float* __restrict__ Agg,
                               int F, int fshift, long long total) {
  long long t = (long long)blockIdx.x * blockDim.x + threadIdx.x;
  if (t >= total) return;
  long long e = t >> fshift;
  int f4 = (int)(t & ((1u << fshift) - 1u)) << 2;
  int r = rows[e], cv = cols[e];
  float nrm = dinv[r] * dinv[cv];
  const unsigned short* src = X + (long long)r * F + f4;
  unsigned int u0 = *(const unsigned int*)(src);
  unsigned int u1 = *(const unsigned int*)(src + 2);
  float* dst = Agg + (long long)cv * F + f4;
  atomicAdd(dst + 0, bf2f((unsigned short)(u0 & 0xFFFFu)) * nrm);
  atomicAdd(dst + 1, bf2f((unsigned short)(u0 >> 16)) * nrm);
  atomicAdd(dst + 2, bf2f((unsigned short)(u1 & 0xFFFFu)) * nrm);
  atomicAdd(dst + 3, bf2f((unsigned short)(u1 >> 16)) * nrm);
}

// scatter-add of f32 messages (conv2): 4 features per thread
__global__ void k_scatter_f32(const int* __restrict__ rows,
                              const int* __restrict__ cols,
                              const float* __restrict__ X,
                              const float* __restrict__ dinv,
                              float* __restrict__ Agg,
                              int F, int fshift, long long total) {
  long long t = (long long)blockIdx.x * blockDim.x + threadIdx.x;
  if (t >= total) return;
  long long e = t >> fshift;
  int f4 = (int)(t & ((1u << fshift) - 1u)) << 2;
  int r = rows[e], cv = cols[e];
  float nrm = dinv[r] * dinv[cv];
  const float* src = X + (long long)r * F + f4;
  float* dst = Agg + (long long)cv * F + f4;
  atomicAdd(dst + 0, src[0] * nrm);
  atomicAdd(dst + 1, src[1] * nrm);
  atomicAdd(dst + 2, src[2] * nrm);
  atomicAdd(dst + 3, src[3] * nrm);
}

// conv1 epilogue: out = relu(agg + xw*dinv^2 + bias)  (out may alias xw)
__global__ void k_post1(const float* __restrict__ agg,
                        const unsigned short* xw,
                        const float* __restrict__ dinv,
                        const float* __restrict__ bias,
                        unsigned short* out, long long n) {
  long long i = (long long)blockIdx.x * blockDim.x + threadIdx.x;
  if (i >= n) return;
  int node = (int)(i >> 8);
  int f    = (int)(i & 255);
  float d = dinv[node];
  float v = agg[i] + bf2f(xw[i]) * d * d + bias[f];
  out[i] = f2bf(fmaxf(v, 0.0f));
}

// conv2 epilogue: out += xw*dinv^2 + bias
__global__ void k_post2(float* __restrict__ out,
                        const float* __restrict__ xw,
                        const float* __restrict__ dinv,
                        const float* __restrict__ bias, long long n) {
  long long i = (long long)blockIdx.x * blockDim.x + threadIdx.x;
  if (i >= n) return;
  int node = (int)(i >> 6);
  int f    = (int)(i & 63);
  float d = dinv[node];
  out[i] = out[i] + xw[i] * d * d + bias[f];
}

// ---------------------------------------------------------------------------
extern "C" void kernel_launch(void* const* d_in, const int* in_sizes, int n_in,
                              void* d_out, int out_size, void* d_ws, size_t ws_size,
                              hipStream_t stream) {
  (void)in_sizes; (void)n_in; (void)out_size; (void)ws_size;

  const float* z   = (const float*)d_in[0];
  const int*   ei  = (const int*)d_in[1];
  const float* W1  = (const float*)d_in[2];
  const float* b1  = (const float*)d_in[3];
  const float* W2  = (const float*)d_in[4];
  const float* b2  = (const float*)d_in[5];
  const float* Wc1 = (const float*)d_in[6];
  const float* bc1 = (const float*)d_in[7];
  const float* Wc2 = (const float*)d_in[8];
  const float* bc2 = (const float*)d_in[9];
  float* out = (float*)d_out;

  const int* rows = ei;                // edge sources  (x_j gather index)
  const int* cols = ei + N_EDGES_;     // edge targets  (aggregation index)

  // ---- workspace layout (256B aligned) ----
  char* ws = (char*)d_ws;
  size_t off = 0;
  auto alloc = [&](size_t bytes) -> void* {
    void* p = ws + off;
    off += (bytes + 255) & ~(size_t)255;
    return p;
  };
  float*          dinv = (float*)alloc((size_t)N_TOTAL_ * 4);
  unsigned short* zb   = (unsigned short*)alloc((size_t)BATCH_ * 64 * 2);
  unsigned short* W1t  = (unsigned short*)alloc((size_t)64 * 256 * 2);
  unsigned short* W2t  = (unsigned short*)alloc((size_t)256 * 768 * 2);
  unsigned short* Wc1t = (unsigned short*)alloc((size_t)64 * 256 * 2);
  unsigned short* Wc2t = (unsigned short*)alloc((size_t)256 * 64 * 2);
  unsigned short* h1   = (unsigned short*)alloc((size_t)BATCH_ * 256 * 2);
  unsigned short* x    = (unsigned short*)alloc((size_t)N_TOTAL_ * 64 * 2);
  unsigned short* xw1  = (unsigned short*)alloc((size_t)N_TOTAL_ * 256 * 2);
  float*          agg1 = (float*)alloc((size_t)N_TOTAL_ * 256 * 4);
  unsigned short* x2   = xw1;          // in-place (index-identical rewrite)
  float*          xw2  = agg1;         // agg1 dead after post1

  // ---- normalization: deg -> dinv ----
  k_init_deg<<<(N_TOTAL_ + 255) / 256, 256, 0, stream>>>(dinv, N_TOTAL_);
  k_deg<<<(N_EDGES_ + 255) / 256, 256, 0, stream>>>(cols, dinv, (long long)N_EDGES_);
  k_dinv<<<(N_TOTAL_ + 255) / 256, 256, 0, stream>>>(dinv, N_TOTAL_);

  // ---- bf16 conversions ----
  k_f32_to_bf16<<<(BATCH_ * 64) / 256, 256, 0, stream>>>(z, zb, (long long)BATCH_ * 64);
  k_transpose_bf16<<<(64 * 256 + 255) / 256, 256, 0, stream>>>(W1, W1t, 64, 256);
  k_transpose_bf16<<<(256 * 768 + 255) / 256, 256, 0, stream>>>(W2, W2t, 256, 768);
  k_transpose_bf16<<<(64 * 256 + 255) / 256, 256, 0, stream>>>(Wc1, Wc1t, 64, 256);
  k_transpose_bf16<<<(256 * 64 + 255) / 256, 256, 0, stream>>>(Wc2, Wc2t, 256, 64);

  // ---- MLP ----
  // h1 = relu(z @ W1 + b1)  : tiles = 2048*16 = 32768, 4 waves/block
  wmma_gemm<<<8192, 128, 0, stream>>>(zb, W1t, b1, nullptr, h1, BATCH_, 256, 64, 1);
  // x = relu(h1 @ W2 + b2)  : tiles = 2048*48 = 98304  (reshape [N,64] is free)
  wmma_gemm<<<24576, 128, 0, stream>>>(h1, W2t, b2, nullptr, x, BATCH_, 768, 256, 1);

  // ---- GCN conv1 ----
  // xw1 = x @ Wc1 : tiles = 24576*16 = 393216
  wmma_gemm<<<98304, 128, 0, stream>>>(x, Wc1t, nullptr, nullptr, xw1, N_TOTAL_, 256, 64, 0);
  hipMemsetAsync(agg1, 0, (size_t)N_TOTAL_ * 256 * 4, stream);
  {
    long long tot = (long long)N_EDGES_ * 64;   // F/4 = 64 per edge
    k_scatter_bf16<<<(unsigned)(tot / 256), 256, 0, stream>>>(
        rows, cols, xw1, dinv, agg1, 256, 6, tot);
  }
  k_post1<<<(unsigned)(((long long)N_TOTAL_ * 256) / 256), 256, 0, stream>>>(
      agg1, xw1, dinv, bc1, x2, (long long)N_TOTAL_ * 256);

  // ---- GCN conv2 ----
  // xw2 = x2 @ Wc2 : tiles = 24576*4 = 98304
  wmma_gemm<<<24576, 128, 0, stream>>>(x2, Wc2t, nullptr, xw2, nullptr, N_TOTAL_, 64, 256, 0);
  hipMemsetAsync(out, 0, (size_t)N_TOTAL_ * 64 * 4, stream);
  {
    long long tot = (long long)N_EDGES_ * 16;   // F/4 = 16 per edge
    k_scatter_f32<<<(unsigned)(tot / 256), 256, 0, stream>>>(
        rows, cols, xw2, dinv, out, 64, 4, tot);
  }
  k_post2<<<(unsigned)(((long long)N_TOTAL_ * 64) / 256), 256, 0, stream>>>(
      out, xw2, dinv, bc2, (long long)N_TOTAL_ * 64);
}
